// Fusion_Module_all_18408229831175
// MI455X (gfx1250) — matrix operand order repaired and checked by reading, not compiled, not run on hardware
//
#include <hip/hip_runtime.h>
#include <hip/hip_bf16.h>
#include <cstdint>
#include <cmath>

typedef __bf16 bf16;
typedef __attribute__((ext_vector_type(16))) __bf16 bf16x16;
typedef __attribute__((ext_vector_type(8)))  __bf16 bf16x8;
typedef __attribute__((ext_vector_type(8)))  float  f32x8;

constexpr int BB = 16;        // batch
constexpr int CC = 768;       // channels
constexpr int NN = 1024;      // tokens per batch (H*W)
constexpr int NT = BB * NN;   // 16384 total tokens
constexpr float LN_EPS = 1e-5f;

union FragU { bf16x16 v; bf16x8 h[2]; };

// ---------------- fragment loaders (wave32, 16x16x32 bf16 WMMA) -------------
// A: 16x32 (MxK). lane m=lane&15 holds K {kb..kb+7} U {kb+16..kb+23}, kb=8*(lane>>4)
static __device__ __forceinline__ bf16x16 load_a(const bf16* base, int lda, int lane) {
    const int row = lane & 15;
    const int kb  = (lane >> 4) << 3;
    const bf16* p = base + (long long)row * lda + kb;
    FragU f;
    f.h[0] = *(const bf16x8*)(p);
    f.h[1] = *(const bf16x8*)(p + 16);
    return f.v;
}
// B: 32x16 (KxN), B[k][n] = src[n*ldb + k]. lane n=lane&15 holds K {kb..kb+7} U {kb+16..kb+23}, kb=8*(lane>>4)
static __device__ __forceinline__ bf16x16 load_b(const bf16* base, int ldb, int lane) {
    const int col = lane & 15;
    const int kb  = (lane >> 4) << 3;
    const bf16* p = base + (long long)col * ldb + kb;
    FragU f;
    f.h[0] = *(const bf16x8*)(p);
    f.h[1] = *(const bf16x8*)(p + 16);
    return f.v;
}

// ---------------- generic batched bf16 GEMM:  Y = A * B^T (+bias) -----------
// Y[m,n] = sum_k A[m*lda+k] * Bsrc[n*ldb+k] + bias[n]
// Block = 256 threads = 8 waves (4 down M x 2 across N); block tile 128x128.
// Each wave computes 32(M) x 64(N): 2 A-frags x 4 B-frags -> 8 WMMAs / k-step.
__global__ __launch_bounds__(256) void gemm_bf16_kernel(
    const bf16* __restrict__ A,  long long sA, int lda,
    const bf16* __restrict__ Bm, long long sB, int ldb,
    float* __restrict__ Yf,  long long sYf,
    bf16*  __restrict__ Ybf, long long sYbf,
    bf16*  __restrict__ YT,  int ldT,          // transposed bf16 out: YT[n*ldT+m]
    const float* __restrict__ bias,
    int K, int ldy)
{
    const int lane = threadIdx.x & 31;
    const int wave = threadIdx.x >> 5;
    const int wm = wave & 3;          // 4 waves down M
    const int wn = wave >> 2;         // 2 waves across N
    const int z  = blockIdx.z;

    const bf16* Ab = A  + (long long)z * sA;
    const bf16* Bb = Bm + (long long)z * sB;
    const int row0 = blockIdx.y * 128 + wm * 32;
    const int col0 = blockIdx.x * 128 + wn * 64;

    f32x8 acc[2][4] = {};
    const bf16* pa = Ab + (long long)row0 * lda;
    const bf16* pb = Bb + (long long)col0 * ldb;

    for (int k = 0; k < K; k += 32) {
        bf16x16 a0 = load_a(pa + k, lda, lane);
        bf16x16 a1 = load_a(pa + (long long)16 * lda + k, lda, lane);
        bf16x16 b0 = load_b(pb + k, ldb, lane);
        bf16x16 b1 = load_b(pb + (long long)16 * ldb + k, ldb, lane);
        bf16x16 b2 = load_b(pb + (long long)32 * ldb + k, ldb, lane);
        bf16x16 b3 = load_b(pb + (long long)48 * ldb + k, ldb, lane);
        __builtin_prefetch(pa + k + 96, 0, 1);   // global_prefetch_b8
        __builtin_prefetch(pb + k + 96, 0, 1);
        acc[0][0] = __builtin_amdgcn_wmma_f32_16x16x32_bf16(false, a0, false, b0, (short)0, acc[0][0], false, false);
        acc[0][1] = __builtin_amdgcn_wmma_f32_16x16x32_bf16(false, a0, false, b1, (short)0, acc[0][1], false, false);
        acc[0][2] = __builtin_amdgcn_wmma_f32_16x16x32_bf16(false, a0, false, b2, (short)0, acc[0][2], false, false);
        acc[0][3] = __builtin_amdgcn_wmma_f32_16x16x32_bf16(false, a0, false, b3, (short)0, acc[0][3], false, false);
        acc[1][0] = __builtin_amdgcn_wmma_f32_16x16x32_bf16(false, a1, false, b0, (short)0, acc[1][0], false, false);
        acc[1][1] = __builtin_amdgcn_wmma_f32_16x16x32_bf16(false, a1, false, b1, (short)0, acc[1][1], false, false);
        acc[1][2] = __builtin_amdgcn_wmma_f32_16x16x32_bf16(false, a1, false, b2, (short)0, acc[1][2], false, false);
        acc[1][3] = __builtin_amdgcn_wmma_f32_16x16x32_bf16(false, a1, false, b3, (short)0, acc[1][3], false, false);
    }

    const int hi  = lane >> 4;
    const int nlo = lane & 15;
    for (int i = 0; i < 2; ++i) {
        const int mrow = row0 + i * 16 + hi * 8;    // rows mrow..mrow+7
        for (int j = 0; j < 4; ++j) {
            const int n = col0 + j * 16 + nlo;
            const float bv = bias ? bias[n] : 0.f;
            float vals[8];
            #pragma unroll
            for (int v = 0; v < 8; ++v) vals[v] = acc[i][j][v] + bv;
            if (Yf) {
                float* y = Yf + (long long)z * sYf;
                #pragma unroll
                for (int v = 0; v < 8; ++v) y[(long long)(mrow + v) * ldy + n] = vals[v];
            }
            if (Ybf) {
                bf16* y = Ybf + (long long)z * sYbf;
                #pragma unroll
                for (int v = 0; v < 8; ++v) y[(long long)(mrow + v) * ldy + n] = (bf16)vals[v];
            }
            if (YT) {
                bf16x8 t8;
                #pragma unroll
                for (int v = 0; v < 8; ++v) t8[v] = (bf16)vals[v];
                *(bf16x8*)(YT + (long long)n * ldT + mrow) = t8;  // 16B aligned
            }
        }
    }
}

// ---------------- fp32 -> bf16 conversion ----------------
__global__ __launch_bounds__(256) void cvt_bf16_kernel(const float* __restrict__ in,
                                                       bf16* __restrict__ out, int n) {
    int i = blockIdx.x * 256 + threadIdx.x;
    if (i < n) out[i] = (bf16)in[i];
}

// ---------------- LayerNorm over C with [B,C,N] -> bf16 [B*N, C] ------------
__global__ __launch_bounds__(256) void ln_tok_kernel(const float* __restrict__ x,
                                                     const float* __restrict__ g,
                                                     const float* __restrict__ be,
                                                     bf16* __restrict__ out) {
    const int b = blockIdx.y;
    const int n = blockIdx.x * 256 + threadIdx.x;
    const float* xb = x + (long long)b * CC * NN + n;   // stride NN over channels
    float s = 0.f, ss = 0.f;
    for (int c = 0; c < CC; ++c) {
        float v = xb[(long long)c * NN];
        s += v; ss += v * v;
    }
    const float mu  = s * (1.f / CC);
    const float var = ss * (1.f / CC) - mu * mu;
    const float rst = rsqrtf(var + LN_EPS);
    bf16* o = out + ((long long)b * NN + n) * CC;
    for (int c = 0; c < CC; ++c) {
        float v = (xb[(long long)c * NN] - mu) * rst * g[c] + be[c];
        o[c] = (bf16)v;
    }
}

// ---------------- softmax over last dim of S [B*N, N], write bf16 P ---------
// Score row (4KB) staged to LDS with one global_load_async_to_lds_b128 / lane.
__global__ __launch_bounds__(256) void softmax_kernel(const float* __restrict__ S,
                                                      bf16* __restrict__ P, float scale) {
    const long long row = (long long)blockIdx.y * NN + blockIdx.x;
    const float* s = S + row * NN;
    __shared__ __align__(16) float srow[NN];
    __shared__ float red[256];
    const int t = threadIdx.x;

    // async-copy the whole row into LDS: lane t covers floats [4t, 4t+4)
    {
        unsigned lds_off = (unsigned)(uintptr_t)(&srow[t * 4]);
        unsigned long long gaddr = (unsigned long long)(s + t * 4);
        asm volatile("global_load_async_to_lds_b128 %0, %1, off"
                     :: "v"(lds_off), "v"(gaddr) : "memory");
        asm volatile("s_wait_asynccnt 0x0" ::: "memory");
    }
    __syncthreads();

    float m = -3.4e38f;
    #pragma unroll
    for (int u = 0; u < NN / 256; ++u) m = fmaxf(m, srow[t + u * 256]);
    red[t] = m; __syncthreads();
    for (int o = 128; o > 0; o >>= 1) { if (t < o) red[t] = fmaxf(red[t], red[t + o]); __syncthreads(); }
    m = red[0]; __syncthreads();
    float e[NN / 256], sum = 0.f;
    #pragma unroll
    for (int u = 0; u < NN / 256; ++u) { e[u] = __expf((srow[t + u * 256] - m) * scale); sum += e[u]; }
    red[t] = sum; __syncthreads();
    for (int o = 128; o > 0; o >>= 1) { if (t < o) red[t] += red[t + o]; __syncthreads(); }
    const float inv = 1.f / red[0];
    bf16* p = P + row * NN;
    #pragma unroll
    for (int u = 0; u < NN / 256; ++u) p[t + u * 256] = (bf16)(e[u] * inv);
}

// ---- x_middle0 = r + t + reorder(mid);  also LN(x_middle0) -> bf16 ---------
// reorder: out[b,n,c] = mid[b, (n*C+c)%N, (n*C+c)/N]
__global__ __launch_bounds__(256) void mid_fuse_ln_kernel(const float* __restrict__ r,
                                                          const float* __restrict__ t,
                                                          const float* __restrict__ mid,
                                                          const float* __restrict__ g,
                                                          const float* __restrict__ be,
                                                          float* __restrict__ xm0,
                                                          bf16* __restrict__ lnout) {
    const long long i = blockIdx.x;          // token 0..NT-1
    const int b = (int)(i / NN);
    const int n = (int)(i % NN);
    const float* rr = r   + i * CC;
    const float* tt = t   + i * CC;
    const float* mb = mid + (long long)b * NN * CC;
    float*       xr = xm0 + i * CC;
    bf16*        lr = lnout + i * CC;
    const int tx = threadIdx.x;
    __shared__ float sh1[256], sh2[256];
    float vals[CC / 256];
    float s = 0.f, ss = 0.f;
    #pragma unroll
    for (int u = 0; u < CC / 256; ++u) {
        const int c = tx + u * 256;
        const long long flat = (long long)n * CC + c;
        const int mrow = (int)(flat & (NN - 1));
        const int mcol = (int)(flat >> 10);            // NN = 1024
        float v = rr[c] + tt[c] + mb[(long long)mrow * CC + mcol];
        vals[u] = v; s += v; ss += v * v;
    }
    sh1[tx] = s; sh2[tx] = ss; __syncthreads();
    for (int o = 128; o > 0; o >>= 1) {
        if (tx < o) { sh1[tx] += sh1[tx + o]; sh2[tx] += sh2[tx + o]; }
        __syncthreads();
    }
    const float mu  = sh1[0] * (1.f / CC);
    const float var = sh2[0] * (1.f / CC) - mu * mu;
    const float rst = rsqrtf(var + LN_EPS);
    #pragma unroll
    for (int u = 0; u < CC / 256; ++u) {
        const int c = tx + u * 256;
        xr[c] = vals[u];
        lr[c] = (bf16)((vals[u] - mu) * rst * g[c] + be[c]);
    }
}

// ---- final: res = (gelu(r+t)+gelu(r*t)) * (xm0+proj), write [B,C,H*W] ------
static __device__ __forceinline__ float gelu_exact(float x) {
    return 0.5f * x * (1.f + erff(x * 0.70710678118654752f));
}
__global__ __launch_bounds__(256) void final_kernel(const float* __restrict__ r,
                                                    const float* __restrict__ t,
                                                    const float* __restrict__ xm0,
                                                    const float* __restrict__ proj,
                                                    float* __restrict__ out) {
    const long long idx = (long long)blockIdx.x * 256 + threadIdx.x;   // over B*N*C
    const int c = (int)(idx % CC);
    const long long bn = idx / CC;
    const int n = (int)(bn % NN);
    const int b = (int)(bn / NN);
    const float rv = r[idx], tv = t[idx];
    const float xs = rv + tv, xu = rv * tv;
    const float xm = xm0[idx] + proj[idx];
    const float res = (gelu_exact(xs) + gelu_exact(xu)) * xm;
    out[((long long)b * CC + c) * NN + n] = res;
}

// ---------------------------------------------------------------------------
extern "C" void kernel_launch(void* const* d_in, const int* in_sizes, int n_in,
                              void* d_out, int out_size, void* d_ws, size_t ws_size,
                              hipStream_t stream) {
    const float* x_rgb = (const float*)d_in[0];
    const float* x_tir = (const float*)d_in[1];
    const float* x_bef = (const float*)d_in[2];
    const float* w0 = (const float*)d_in[3];
    const float* b0 = (const float*)d_in[4];
    const float* w1 = (const float*)d_in[5];
    const float* b1 = (const float*)d_in[6];
    const float* w2 = (const float*)d_in[7];
    const float* b2 = (const float*)d_in[8];
    const float* ln_g = (const float*)d_in[9];
    const float* ln_b = (const float*)d_in[10];
    const float* wp = (const float*)d_in[11];
    const float* bp = (const float*)d_in[12];
    float* out = (float*)d_out;

    char* ws = (char*)d_ws;
    const size_t A_BF  = (size_t)NT * CC * 2;        // 25165824 bf16 token buf
    const size_t A_F32 = (size_t)NT * CC * 4;        // 50331648
    const size_t S_F32 = (size_t)BB * NN * NN * 4;   // 67108864
    const size_t P_BF  = (size_t)BB * NN * NN * 2;   // 33554432

    bf16* Aln0 = (bf16*)(ws + 0);
    bf16* Aln1 = (bf16*)(ws + A_BF);
    bf16* Aln2 = (bf16*)(ws + 2 * A_BF);
    bf16* Pbuf = (bf16*)(ws + 0);                    // reuse (Aln dead)
    bf16* lnm  = (bf16*)(ws + P_BF);                 // reuse (fits in Aln region)
    float* rf  = (float*)(ws + 3 * A_BF);
    float* tf  = (float*)(ws + 3 * A_BF + A_F32);
    bf16* rbf  = (bf16*)(ws + 3 * A_BF + 2 * A_F32);
    bf16* fbf  = (bf16*)(ws + 4 * A_BF + 2 * A_F32);
    bf16* tT   = (bf16*)(ws + 5 * A_BF + 2 * A_F32);
    float* proj = (float*)(ws + 3 * A_BF + 2 * A_F32);   // reuse rbf/fbf space
    float* Sbuf = (float*)(ws + 6 * A_BF + 2 * A_F32);
    float* mid  = (float*)(ws + 6 * A_BF + 2 * A_F32);   // reuse S space
    float* xm0  = (float*)(ws + 6 * A_BF + 2 * A_F32 + S_F32);
    bf16* wbf[4];
    {
        size_t off = 6 * A_BF + 2 * A_F32 + S_F32 + A_F32;
        for (int i = 0; i < 4; ++i) { wbf[i] = (bf16*)(ws + off); off += (size_t)CC * CC * 2; }
    }

    const int WN = CC * CC;
    cvt_bf16_kernel<<<(WN + 255) / 256, 256, 0, stream>>>(w0, wbf[0], WN);
    cvt_bf16_kernel<<<(WN + 255) / 256, 256, 0, stream>>>(w1, wbf[1], WN);
    cvt_bf16_kernel<<<(WN + 255) / 256, 256, 0, stream>>>(w2, wbf[2], WN);
    cvt_bf16_kernel<<<(WN + 255) / 256, 256, 0, stream>>>(wp, wbf[3], WN);

    dim3 lnGrid(NN / 256, BB);
    ln_tok_kernel<<<lnGrid, 256, 0, stream>>>(x_rgb, ln_g, ln_b, Aln0);
    ln_tok_kernel<<<lnGrid, 256, 0, stream>>>(x_tir, ln_g, ln_b, Aln1);
    ln_tok_kernel<<<lnGrid, 256, 0, stream>>>(x_bef, ln_g, ln_b, Aln2);

    // r = LN(rgb) @ w0^T + b0   (fp32 + bf16)
    dim3 gemmGrid(CC / 128, NT / 128, 1);
    gemm_bf16_kernel<<<gemmGrid, 256, 0, stream>>>(Aln0, 0, CC, wbf[0], 0, CC,
                                                   rf, 0, rbf, 0, nullptr, 0, b0, CC, CC);
    // t = LN(tir) @ w1^T + b1   (fp32 + transposed bf16 [C, NT])
    gemm_bf16_kernel<<<gemmGrid, 256, 0, stream>>>(Aln1, 0, CC, wbf[1], 0, CC,
                                                   tf, 0, nullptr, 0, tT, NT, b1, CC, CC);
    // f = LN(before) @ w2^T + b2 (bf16 only)
    gemm_bf16_kernel<<<gemmGrid, 256, 0, stream>>>(Aln2, 0, CC, wbf[2], 0, CC,
                                                   nullptr, 0, fbf, 0, nullptr, 0, b2, CC, CC);
    // S[b] = r[b] @ f[b]^T  (fp32, batched)
    dim3 sGrid(NN / 128, NN / 128, BB);
    gemm_bf16_kernel<<<sGrid, 256, 0, stream>>>(rbf, (long long)NN * CC, CC,
                                                fbf, (long long)NN * CC, CC,
                                                Sbuf, (long long)NN * NN,
                                                nullptr, 0, nullptr, 0, nullptr, CC, NN);
    // P = softmax(S * C^-0.5) -> bf16
    dim3 smGrid(NN, BB);
    softmax_kernel<<<smGrid, 256, 0, stream>>>(Sbuf, Pbuf, 1.f / sqrtf((float)CC));
    // mid[b] = P[b] @ t[b]   (B frag from tT: tT[c*NT + b*NN + m])
    dim3 mGrid(CC / 128, NN / 128, BB);
    gemm_bf16_kernel<<<mGrid, 256, 0, stream>>>(Pbuf, (long long)NN * NN, NN,
                                                tT, (long long)NN, NT,
                                                mid, (long long)NN * CC,
                                                nullptr, 0, nullptr, 0, nullptr, NN, CC);
    // x_middle0 = r + t + reorder(mid); lnm = LN(x_middle0)
    mid_fuse_ln_kernel<<<NT, 256, 0, stream>>>(rf, tf, mid, ln_g, ln_b, xm0, lnm);
    // proj = lnm @ wp^T + bp
    gemm_bf16_kernel<<<gemmGrid, 256, 0, stream>>>(lnm, 0, CC, wbf[3], 0, CC,
                                                   proj, 0, nullptr, 0, nullptr, 0, bp, CC, CC);
    // res -> [B, C, H, W]
    final_kernel<<<(int)(((long long)NT * CC) / 256), 256, 0, stream>>>(rf, tf, xm0, proj, out);
}
